// GNNNet_22454089023916
// MI455X (gfx1250) — compile-verified, matrix operand order checked by reading, not run on hardware
//
#include <hip/hip_runtime.h>

// ---------------------------------------------------------------------------
// GNNNet (dual-branch GCN) for MI455X / gfx1250.
// Dense X*W stages: v_wmma_f32_16x16x32_f16 (f16 in, f32 accumulate), with a
// register double-buffered pipeline: next tile's global loads are issued
// before the current tile's WMMA work, hiding HBM/L2 latency behind math.
// Fragments come from LDS as ds_load_b128 pairs (A row-major, B transposed).
// Edge aggregation uses global_atomic_add_f32; workload is HBM/L2 bound.
// ---------------------------------------------------------------------------

typedef __attribute__((ext_vector_type(16))) _Float16 v16h;
typedef __attribute__((ext_vector_type(8)))  _Float16 v8h;
typedef __attribute__((ext_vector_type(2)))  _Float16 v2h;
typedef __attribute__((ext_vector_type(8)))  float    v8f;

#define GEMM_THREADS 128   // 4 waves (wave32)
#define TM 128             // block tile M  (32 rows per wave)
#define TN 64              // block tile N
#define TK 32              // K chunk (one WMMA K)
#define LSTR 40            // LDS row stride in halves (80 B: 16B-aligned, non-pow2 banks)
#define A_ITERS (TM * TK / 2 / GEMM_THREADS)   // 16 float2 per thread
#define B_ITERS (TK / 2 * TN / GEMM_THREADS)   // 8  float2 per thread

// Build a v16h fragment from a fragment-ordered LDS row.
// Lane's elements are two contiguous 8-half runs at k = lhalf*8 and 16+lhalf*8.
static __device__ inline v16h load_frag_row(const _Float16* rowp, int lhalf)
{
    v8h lo = *(const v8h*)(rowp + (lhalf << 3));        // ds_load_b128
    v8h hi = *(const v8h*)(rowp + 16 + (lhalf << 3));   // ds_load_b128
    return __builtin_shufflevector(lo, hi, 0, 1, 2, 3, 4, 5, 6, 7,
                                           8, 9, 10, 11, 12, 13, 14, 15);
}

// ---- A tile: registers <- global ------------------------------------------
static __device__ inline void load_A_regs(const float* __restrict__ A, int M, int K,
                                          int blockM, int k0, int tid, float2* areg)
{
    const bool full = ((K & 1) == 0) && (blockM + TM <= M) && (k0 + TK <= K);
    if (full) {
        #pragma unroll
        for (int it = 0; it < A_ITERS; ++it) {
            int i = tid + it * GEMM_THREADS;
            int r = i >> 4;                      // 16 pairs per row
            int p = (i & 15) << 1;
            areg[it] = *(const float2*)(A + (size_t)(blockM + r) * K + k0 + p);
        }
    } else {
        #pragma unroll
        for (int it = 0; it < A_ITERS; ++it) {
            int i  = tid + it * GEMM_THREADS;
            int r  = i >> 4;
            int p  = (i & 15) << 1;
            int gm = blockM + r, gk = k0 + p;
            float2 v; v.x = 0.0f; v.y = 0.0f;
            if (gm < M && gk < K)     v.x = A[(size_t)gm * K + gk];
            if (gm < M && gk + 1 < K) v.y = A[(size_t)gm * K + gk + 1];
            areg[it] = v;
        }
    }
}

static __device__ inline void store_A_lds(_Float16 (*As)[LSTR], int tid, const float2* areg)
{
    #pragma unroll
    for (int it = 0; it < A_ITERS; ++it) {
        int i = tid + it * GEMM_THREADS;
        int r = i >> 4;
        int p = (i & 15) << 1;
        v2h h; h[0] = (_Float16)areg[it].x; h[1] = (_Float16)areg[it].y;
        *(v2h*)&As[r][p] = h;                    // ds_store_b32
    }
}

// ---- B tile (transposed in LDS): registers <- global -----------------------
static __device__ inline void load_B_regs(const float* __restrict__ B, int K, int Nc,
                                          int blockN, int k0, int tid, float2* breg)
{
    const bool full = (k0 + TK <= K) && (blockN + TN <= Nc);
    if (full) {
        #pragma unroll
        for (int it = 0; it < B_ITERS; ++it) {
            int i  = tid + it * GEMM_THREADS;
            int r2 = (i >> 6) << 1;              // even k within chunk
            int c  = i & (TN - 1);
            const float* gp = B + (size_t)(k0 + r2) * Nc + blockN + c;
            float2 v; v.x = gp[0]; v.y = gp[Nc];
            breg[it] = v;
        }
    } else {
        #pragma unroll
        for (int it = 0; it < B_ITERS; ++it) {
            int i  = tid + it * GEMM_THREADS;
            int r2 = (i >> 6) << 1;
            int c  = i & (TN - 1);
            int gk = k0 + r2, gn = blockN + c;
            float2 v; v.x = 0.0f; v.y = 0.0f;
            if (gk < K && gn < Nc)     v.x = B[(size_t)gk * Nc + gn];
            if (gk + 1 < K && gn < Nc) v.y = B[(size_t)(gk + 1) * Nc + gn];
            breg[it] = v;
        }
    }
}

static __device__ inline void store_B_lds(_Float16 (*Bt)[LSTR], int tid, const float2* breg)
{
    #pragma unroll
    for (int it = 0; it < B_ITERS; ++it) {
        int i  = tid + it * GEMM_THREADS;
        int r2 = (i >> 6) << 1;
        int c  = i & (TN - 1);
        v2h h; h[0] = (_Float16)breg[it].x; h[1] = (_Float16)breg[it].y;
        *(v2h*)&Bt[c][r2] = h;                   // ds_store_b32
    }
}

// ---------------------------------------------------------------------------
// WMMA GEMM: C[M,Nc] = A[M,K] (row-major f32) * B[K,Nc] (row-major f32)
// optionally + bias[Nc], optionally ReLU.
// ---------------------------------------------------------------------------
__global__ __launch_bounds__(GEMM_THREADS)
void wmma_gemm_kernel(const float* __restrict__ A, const float* __restrict__ B,
                      const float* __restrict__ bias, float* __restrict__ C,
                      int M, int Nc, int K, int relu)
{
    __shared__ __align__(16) _Float16 As[TM][LSTR];   // 128 x 32 (padded)
    __shared__ __align__(16) _Float16 Bt[TN][LSTR];   // 64 x 32, B transposed

    const int tid    = threadIdx.x;
    const int wave   = tid >> 5;         // 0..3
    const int lane   = tid & 31;
    const int lmod   = lane & 15;
    const int lhalf  = lane >> 4;        // 0/1
    const int blockM = blockIdx.y * TM;
    const int blockN = blockIdx.x * TN;

    v8f acc[2][4] = {};                  // wave's 32x64 stripe: 2 (M) x 4 (N) tiles
    float2 areg[A_ITERS];
    float2 breg[B_ITERS];

    // prologue: prefetch tile 0 into registers
    load_A_regs(A, M, K, blockM, 0, tid, areg);
    load_B_regs(B, K, Nc, blockN, 0, tid, breg);

    for (int k0 = 0; k0 < K; k0 += TK) {
        // drain prefetched registers into LDS
        store_A_lds(As, tid, areg);
        store_B_lds(Bt, tid, breg);
        __syncthreads();

        // issue next tile's global loads; they complete during the WMMA work
        if (k0 + TK < K) {
            load_A_regs(A, M, K, blockM, k0 + TK, tid, areg);
            load_B_regs(B, K, Nc, blockN, k0 + TK, tid, breg);
        }

        // fragments: 2 ds_load_b128 each
        v16h afrag[2];
        #pragma unroll
        for (int mi = 0; mi < 2; ++mi)
            afrag[mi] = load_frag_row(&As[wave * 32 + mi * 16 + lmod][0], lhalf);

        #pragma unroll
        for (int j = 0; j < 4; ++j) {
            v16h bfrag = load_frag_row(&Bt[j * 16 + lmod][0], lhalf);
            #pragma unroll
            for (int mi = 0; mi < 2; ++mi)
                acc[mi][j] = __builtin_amdgcn_wmma_f32_16x16x32_f16(
                    false, afrag[mi], false, bfrag, (short)0, acc[mi][j],
                    false, false);
        }
        __syncthreads();
    }

    // store: C/D layout — VGPR i holds m = i + 8*lhalf, lane lmod holds n
    #pragma unroll
    for (int j = 0; j < 4; ++j) {
        int n = blockN + j * 16 + lmod;
        if (n >= Nc) continue;
        float bv = bias ? bias[n] : 0.0f;
        #pragma unroll
        for (int mi = 0; mi < 2; ++mi) {
            #pragma unroll
            for (int i = 0; i < 8; ++i) {
                int m = blockM + wave * 32 + mi * 16 + i + (lhalf << 3);
                if (m < M) {
                    float v = acc[mi][j][i] + bv;
                    if (relu) v = fmaxf(v, 0.0f);
                    C[(size_t)m * Nc + n] = v;
                }
            }
        }
    }
}

// ---------------------------------------------------------------------------
// elementwise / graph kernels
// ---------------------------------------------------------------------------
__global__ void fill_kernel(float* __restrict__ p, float v, size_t n)
{
    size_t i = (size_t)blockIdx.x * blockDim.x + threadIdx.x;
    size_t s = (size_t)gridDim.x * blockDim.x;
    for (; i < n; i += s) p[i] = v;
}

__global__ void deg_count_kernel(const int* __restrict__ col, float* __restrict__ deg, int E)
{
    int i = blockIdx.x * blockDim.x + threadIdx.x;
    if (i < E) atomicAdd(&deg[col[i]], 1.0f);
}

__global__ void rsqrt_kernel(float* __restrict__ d, int N)
{
    int i = blockIdx.x * blockDim.x + threadIdx.x;
    if (i < N) d[i] = rsqrtf(d[i]);   // deg >= 1 always (self loops)
}

// out[col[e], :] += h[row[e], :] * dinv[row]*dinv[col]
__global__ void scatter_edges_kernel(const float* __restrict__ h,
                                     const int* __restrict__ row,
                                     const int* __restrict__ col,
                                     const float* __restrict__ dinv,
                                     float* __restrict__ out, int E, int F)
{
    int e = blockIdx.x;
    if (e >= E) return;
    int r = row[e], c = col[e];
    float norm = dinv[r] * dinv[c];
    const float* hr = h + (size_t)r * F;
    float*       oc = out + (size_t)c * F;
    __builtin_prefetch(hr, 0, 1);      // global_prefetch_b8
    for (int f = threadIdx.x; f < F; f += blockDim.x)
        atomicAdd(&oc[f], hr[f] * norm);
}

// out = relu(out + h * dinv^2 (self loop) + bias)
__global__ void selfloop_bias_relu_kernel(const float* __restrict__ h,
                                          const float* __restrict__ dinv,
                                          const float* __restrict__ b,
                                          float* __restrict__ out, int N, int F)
{
    size_t total = (size_t)N * F;
    size_t i = (size_t)blockIdx.x * blockDim.x + threadIdx.x;
    size_t s = (size_t)gridDim.x * blockDim.x;
    for (; i < total; i += s) {
        size_t node = i / F;
        int    f    = (int)(i % F);
        float  dv   = dinv[node];
        float  v    = out[i] + h[i] * dv * dv + b[f];
        out[i] = fmaxf(v, 0.0f);
    }
}

__global__ void pool_sum_kernel(const float* __restrict__ x, const int* __restrict__ batch,
                                float* __restrict__ pooled, float* __restrict__ cnt,
                                int N, int F)
{
    int node = blockIdx.x;
    if (node >= N) return;
    int g = batch[node];
    if (threadIdx.x == 0) atomicAdd(&cnt[g], 1.0f);
    const float* xr = x + (size_t)node * F;
    float*       pg = pooled + (size_t)g * F;
    for (int f = threadIdx.x; f < F; f += blockDim.x)
        atomicAdd(&pg[f], xr[f]);
}

__global__ void pool_div_kernel(float* __restrict__ pooled, const float* __restrict__ cnt,
                                int G, int F)
{
    int i = blockIdx.x * blockDim.x + threadIdx.x;
    if (i < G * F) pooled[i] /= fmaxf(cnt[i / F], 1.0f);
}

// ---------------------------------------------------------------------------
// host-side orchestration
// ---------------------------------------------------------------------------
static inline void launch_gemm(const float* A, const float* B, const float* bias, float* C,
                               int M, int Nc, int K, int relu, hipStream_t s)
{
    dim3 g((Nc + TN - 1) / TN, (M + TM - 1) / TM);
    wmma_gemm_kernel<<<g, GEMM_THREADS, 0, s>>>(A, B, bias, C, M, Nc, K, relu);
}

static inline void launch_fill(float* p, float v, size_t n, hipStream_t s)
{
    unsigned blocks = (unsigned)((n + 255) / 256);
    fill_kernel<<<blocks, 256, 0, s>>>(p, v, n);
}

static void run_branch(const float* x, const int* adj, const int* batch,
                       int N, int E, int F0, int F1, int F2, int F3,
                       const float* W1, const float* b1,
                       const float* W2, const float* b2,
                       const float* W3, const float* b3,
                       const float* fc1W, const float* fc1b,
                       const float* fc2W, const float* fc2b,
                       float* outp,
                       float* bufH, float* bufX, float* dinv,
                       float* pooled, float* cnt, float* fc1out,
                       hipStream_t stream)
{
    const int G = 1024, HID = 1024, OUT = 160;
    const int* row = adj;
    const int* col = adj + E;

    // symmetric norm with self loops: deg[c] = incoming + 1 ; dinv = rsqrt(deg)
    launch_fill(dinv, 1.0f, (size_t)N, stream);
    deg_count_kernel<<<(E + 255) / 256, 256, 0, stream>>>(col, dinv, E);
    rsqrt_kernel<<<(N + 255) / 256, 256, 0, stream>>>(dinv, N);

    // ---- conv layer 1: F0 -> F1
    launch_gemm(x, W1, nullptr, bufH, N, F1, F0, 0, stream);
    launch_fill(bufX, 0.0f, (size_t)N * F1, stream);
    scatter_edges_kernel<<<E, 128, 0, stream>>>(bufH, row, col, dinv, bufX, E, F1);
    selfloop_bias_relu_kernel<<<(unsigned)(((size_t)N * F1 + 255) / 256), 256, 0, stream>>>(
        bufH, dinv, b1, bufX, N, F1);

    // ---- conv layer 2: F1 -> F2
    launch_gemm(bufX, W2, nullptr, bufH, N, F2, F1, 0, stream);
    launch_fill(bufX, 0.0f, (size_t)N * F2, stream);
    scatter_edges_kernel<<<E, 128, 0, stream>>>(bufH, row, col, dinv, bufX, E, F2);
    selfloop_bias_relu_kernel<<<(unsigned)(((size_t)N * F2 + 255) / 256), 256, 0, stream>>>(
        bufH, dinv, b2, bufX, N, F2);

    // ---- conv layer 3: F2 -> F3
    launch_gemm(bufX, W3, nullptr, bufH, N, F3, F2, 0, stream);
    launch_fill(bufX, 0.0f, (size_t)N * F3, stream);
    scatter_edges_kernel<<<E, 128, 0, stream>>>(bufH, row, col, dinv, bufX, E, F3);
    selfloop_bias_relu_kernel<<<(unsigned)(((size_t)N * F3 + 255) / 256), 256, 0, stream>>>(
        bufH, dinv, b3, bufX, N, F3);

    // ---- global mean pool -> [G, F3]
    launch_fill(pooled, 0.0f, (size_t)G * F3, stream);
    launch_fill(cnt, 0.0f, (size_t)G, stream);
    pool_sum_kernel<<<N, 128, 0, stream>>>(bufX, batch, pooled, cnt, N, F3);
    pool_div_kernel<<<(G * F3 + 255) / 256, 256, 0, stream>>>(pooled, cnt, G, F3);

    // ---- FC1 (relu) and FC2 (into final output)
    launch_gemm(pooled, fc1W, fc1b, fc1out, G, HID, F3, 1, stream);
    launch_gemm(fc1out, fc2W, fc2b, outp, G, OUT, HID, 0, stream);
}

extern "C" void kernel_launch(void* const* d_in, const int* in_sizes, int n_in,
                              void* d_out, int out_size, void* d_ws, size_t ws_size,
                              hipStream_t stream)
{
    (void)out_size; (void)ws_size;
    if (n_in < 26) return;

    const float* drug_x   = (const float*)d_in[0];
    const int*   drug_adj = (const int*)  d_in[1];
    const int*   ibatch   = (const int*)  d_in[2];
    const float* pro_x    = (const float*)d_in[3];
    const int*   pro_adj  = (const int*)  d_in[4];
    const int*   pbatch   = (const int*)  d_in[5];

    const float* P[20];
    for (int i = 0; i < 20; ++i) P[i] = (const float*)d_in[6 + i];
    // P: 0 mol_W1, 1 mol_b1, 2 mol_W2, 3 mol_b2, 4 mol_W3, 5 mol_b3,
    //    6 mol_fc1_W, 7 mol_fc1_b, 8 mol_fc2_W, 9 mol_fc2_b, 10.. same for pro

    const int Fm = 78, Fp = 54;
    const int Nm = in_sizes[0] / Fm;
    const int Em = in_sizes[1] / 2;
    const int Np = in_sizes[3] / Fp;
    const int Ep = in_sizes[4] / 2;

    // workspace layout (floats)
    float* ws = (float*)d_ws;
    const size_t BUF = (size_t)100000 * 216;          // max N*F3 across branches
    float* bufH   = ws;
    float* bufX   = bufH + BUF;
    float* dinv   = bufX + BUF;
    float* pooled = dinv + 100000;
    float* cnt    = pooled + (size_t)1024 * 312;
    float* fc1out = cnt + 1024;

    float* out = (float*)d_out;                       // [2048, 160]

    // mol branch: 78 -> 78 -> 156 -> 312
    run_branch(drug_x, drug_adj, ibatch, Nm, Em, Fm, Fm, 2 * Fm, 4 * Fm,
               P[0], P[1], P[2], P[3], P[4], P[5], P[6], P[7], P[8], P[9],
               out, bufH, bufX, dinv, pooled, cnt, fc1out, stream);

    // pro branch: 54 -> 54 -> 108 -> 216
    run_branch(pro_x, pro_adj, pbatch, Np, Ep, Fp, Fp, 2 * Fp, 4 * Fp,
               P[10], P[11], P[12], P[13], P[14], P[15], P[16], P[17], P[18], P[19],
               out + (size_t)1024 * 160, bufH, bufX, dinv, pooled, cnt, fc1out, stream);
}
